// TransformerEncoder3D_4930622456464
// MI455X (gfx1250) — matrix operand order repaired and checked by reading, not compiled, not run on hardware
//
#include <hip/hip_runtime.h>
#include <hip/hip_bf16.h>

typedef _Float16 f16;
typedef __attribute__((ext_vector_type(16))) _Float16 v16h;
typedef __attribute__((ext_vector_type(8)))  _Float16 v8h;
typedef __attribute__((ext_vector_type(8)))  float    v8f;

#define EMB   128
#define ZDIM  32
#define XY    4096          // 64*64 heads
#define NTOK  131072        // ZDIM*XY tokens
#define SCALE 11.313708498984760f   // sqrt(128)

static __device__ inline v8f wmma16(v16h a, v16h b, v8f c) {
    return __builtin_amdgcn_wmma_f32_16x16x32_f16(false, a, false, b, (short)0, c, false, false);
}

// ---------------------------------------------------------------------------
// Weight pre-format: W[K,Ncols] f32 (row major) -> per-lane WMMA B-fragment
// layout: Wfmt[((ntile*(K/32)+s)*32 + lane)*16 + j] = W[32s + 8*(lane>>4) + (j&7) + 16*(j>>3)][16*ntile + (lane&15)]
// qkvperm=1 additionally de-interleaves qkv columns (src col = d*3 + g).
// ---------------------------------------------------------------------------
__global__ void preformat_w(const float* __restrict__ W, f16* __restrict__ Wfmt,
                            int K, int Ncols, int qkvperm) {
    int f = blockIdx.x * 256 + threadIdx.x;
    int total = K * Ncols;
    if (f >= total) return;
    int j  = f & 15;
    int L  = (f >> 4) & 31;
    int ks = K >> 5;
    int s  = (f >> 9) % ks;
    int tile = f / (512 * ks);
    int k = 32 * s + 8 * (L >> 4) + (j & 7) + 16 * (j >> 3);
    int n = 16 * tile + (L & 15);
    int srccol = qkvperm ? ((n & 127) * 3 + (n >> 7)) : n;
    Wfmt[f] = (f16)W[(size_t)k * Ncols + srccol];
}

__global__ void permute_bqkv(const float* __restrict__ b, float* __restrict__ bq) {
    int c = blockIdx.x * 128 + threadIdx.x;
    if (c < 384) bq[c] = b[(c & 127) * 3 + (c >> 7)];
}

// ---------------------------------------------------------------------------
// InstanceNorm over the token axis: partial sums per channel (coalesced rows).
// ---------------------------------------------------------------------------
__global__ void zero_stats(float* st) { st[threadIdx.x] = 0.0f; }

__global__ void stats_kernel(const float* __restrict__ x, float* __restrict__ st) {
    int c = threadIdx.x;                       // 128 threads = channels
    size_t row0 = (size_t)blockIdx.x * 128;    // 1024 blocks * 128 rows
    float s1 = 0.f, s2 = 0.f;
    for (int r = 0; r < 128; ++r) {
        float v = x[(row0 + r) * EMB + c];
        s1 += v; s2 += v * v;
    }
    atomicAdd(&st[c], s1);
    atomicAdd(&st[EMB + c], s2);
}

__global__ void normcast_kernel(const float* __restrict__ x, const float* __restrict__ st,
                                const float* __restrict__ g, const float* __restrict__ b,
                                f16* __restrict__ xn) {
    size_t i = (size_t)blockIdx.x * 256 + threadIdx.x;
    int c = (int)(i & 127);
    float mu  = st[c] * (1.0f / (float)NTOK);
    float var = st[EMB + c] * (1.0f / (float)NTOK) - mu * mu;
    float v = (x[i] - mu) * rsqrtf(var + 1e-5f) * g[c] + b[c];
    xn[i] = (f16)v;
}

// ---------------------------------------------------------------------------
// WMMA GEMM: C[M,Ncols] = A[M,K](f16) x Wfmt(pre-formatted f16) + bias
// 8 waves/block, each wave does a 16x64 strip (A fragment reused 4x).
// ACT: 0 = none, 1 = hard_swish.  RESID: 0 -> store f16, 1 -> f32 += (residual).
// ---------------------------------------------------------------------------
template<int ACT, int RESID>
__global__ __launch_bounds__(256)
void gemm_kernel(const f16* __restrict__ A, const f16* __restrict__ Wfmt,
                 const float* __restrict__ bias, f16* __restrict__ outH,
                 float* __restrict__ outF, int K, int Ncols) {
    const int lane = threadIdx.x & 31;
    const int wid  = threadIdx.x >> 5;
    const int half = lane >> 4;
    const int l15  = lane & 15;
    const int ngroups = Ncols >> 6;
    const int gw    = blockIdx.x * 8 + wid;
    const int mtile = gw / ngroups;
    const int ngrp  = gw % ngroups;
    const int ks = K >> 5;

    const f16* arow = A + (size_t)(mtile * 16 + l15) * K;
    v8f acc[4] = {v8f{}, v8f{}, v8f{}, v8f{}};

    for (int s = 0; s < ks; ++s) {
        v16h a;
        const f16* ap = arow + s * 32 + half * 8;
        *(v8h*)&a        = *(const v8h*)(ap);
        *((v8h*)&a + 1)  = *(const v8h*)(ap + 16);
#pragma unroll
        for (int t = 0; t < 4; ++t) {
            int ntile = ngrp * 4 + t;
            const v16h* bp = (const v16h*)(Wfmt + ((size_t)(ntile * ks + s) * 32 + lane) * 16);
            acc[t] = wmma16(a, *bp, acc[t]);
        }
    }
#pragma unroll
    for (int t = 0; t < 4; ++t) {
        int col = (ngrp * 4 + t) * 16 + l15;
        float bv = bias[col];
#pragma unroll
        for (int r = 0; r < 8; ++r) {
            int row = mtile * 16 + r + half * 8;
            float c = acc[t][r] + bv;
            if (ACT == 1) {
                float t6 = fminf(fmaxf(c + 3.0f, 0.0f), 6.0f);
                c = c * t6 * (1.0f / 6.0f);
            }
            if (RESID) outF[(size_t)row * Ncols + col] += c;
            else       outH[(size_t)row * Ncols + col] = (f16)c;
        }
    }
}

// ---------------------------------------------------------------------------
// Attention: 1 wave per head (4096 heads, Z=32, E=128).
// energy = q k^T (WMMA, K=128) -> softmax/scale/gauss in registers ->
// att staged in LDS (16B-aligned rows) -> out = att x v (WMMA, K=32).
// qkv layout: [token n][384] with cols 0..127=q, 128..255=k, 256..383=v,
// token n = z*XY + head.
// ---------------------------------------------------------------------------
__global__ __launch_bounds__(256)
void attn_kernel(const f16* __restrict__ qkv, f16* __restrict__ outA) {
    __shared__ f16 satt[8][32][40];            // padded: 80B row stride, 16B-aligned
    const int lane = threadIdx.x & 31;
    const int wid  = threadIdx.x >> 5;
    const int h    = blockIdx.x * 8 + wid;
    const int half = lane >> 4;
    const int l15  = lane & 15;

    // ---- energy = q k^T  (2x2 tiles of 16x16, K=128) ----
    v8f e[2][2] = {{v8f{}, v8f{}}, {v8f{}, v8f{}}};
    for (int s = 0; s < 4; ++s) {
        v16h a[2];
#pragma unroll
        for (int mt = 0; mt < 2; ++mt) {
            int z = mt * 16 + l15;
            const f16* ap = qkv + ((size_t)z * XY + h) * 384 + s * 32 + half * 16;
            *(v8h*)&a[mt]       = *(const v8h*)ap;
            *((v8h*)&a[mt] + 1) = *(const v8h*)(ap + 8);
        }
#pragma unroll
        for (int nt = 0; nt < 2; ++nt) {
            int zp = nt * 16 + l15;
            const v16h* bp = (const v16h*)(qkv + ((size_t)zp * XY + h) * 384 + 128 + s * 32 + half * 16);
            v16h b = *bp;
            e[0][nt] = wmma16(a[0], b, e[0][nt]);
            e[1][nt] = wmma16(a[1], b, e[1][nt]);
        }
    }

    // ---- softmax over 32 cols, then /SCALE, then gaussian mask ----
#pragma unroll
    for (int mt = 0; mt < 2; ++mt) {
#pragma unroll
        for (int r = 0; r < 8; ++r) {
            float x0 = e[mt][0][r], x1 = e[mt][1][r];
            float m = fmaxf(x0, x1);
#pragma unroll
            for (int off = 1; off < 16; off <<= 1) m = fmaxf(m, __shfl_xor(m, off, 32));
            float p0 = __expf(x0 - m), p1 = __expf(x1 - m);
            float su = p0 + p1;
#pragma unroll
            for (int off = 1; off < 16; off <<= 1) su += __shfl_xor(su, off, 32);
            float inv = 1.0f / (su * SCALE);
            int row = mt * 16 + r + half * 8;
            float d0 = (float)(l15 - row);
            float d1 = (float)(16 + l15 - row);
            satt[wid][row][l15]      = (f16)(p0 * inv * __expf(-d0 * d0 * (1.0f / 16.0f)));
            satt[wid][row][16 + l15] = (f16)(p1 * inv * __expf(-d1 * d1 * (1.0f / 16.0f)));
        }
    }
    __syncthreads();

    // ---- out = att x v (K=32, 8 column tiles of E) ----
    v16h a2[2];
#pragma unroll
    for (int mt = 0; mt < 2; ++mt) {
        const f16* ap = &satt[wid][mt * 16 + l15][half * 8];
        *(v8h*)&a2[mt]       = *(const v8h*)ap;
        *((v8h*)&a2[mt] + 1) = *(const v8h*)(ap + 16);
    }
    for (int nt = 0; nt < 8; ++nt) {
        v16h b;
#pragma unroll
        for (int j = 0; j < 16; ++j) {
            int zp = half * 8 + (j & 7) + 16 * (j >> 3);
            b[j] = qkv[((size_t)zp * XY + h) * 384 + 256 + nt * 16 + l15];
        }
#pragma unroll
        for (int mt = 0; mt < 2; ++mt) {
            v8f o = {};
            o = wmma16(a2[mt], b, o);
#pragma unroll
            for (int r = 0; r < 8; ++r) {
                int z = mt * 16 + r + half * 8;
                outA[((size_t)z * XY + h) * EMB + nt * 16 + l15] = (f16)o[r];
            }
        }
    }
}

// ---------------------------------------------------------------------------
// Layout transposes (img [128, N] <-> act [N, 128]) with LDS tiling.
// ---------------------------------------------------------------------------
__global__ __launch_bounds__(256) void transpose_in(const float* __restrict__ img,
                                                    float* __restrict__ x) {
    __shared__ float tile[32][33];
    int nb = blockIdx.x, mb = blockIdx.y;
    int tx = threadIdx.x & 31, ty = threadIdx.x >> 5;
#pragma unroll
    for (int i = 0; i < 4; ++i)
        tile[ty + i * 8][tx] = img[(size_t)(mb * 32 + ty + i * 8) * NTOK + nb * 32 + tx];
    __syncthreads();
#pragma unroll
    for (int i = 0; i < 4; ++i)
        x[(size_t)(nb * 32 + ty + i * 8) * EMB + mb * 32 + tx] = tile[tx][ty + i * 8];
}

__global__ __launch_bounds__(256) void final_out(const float* __restrict__ x,
                                                 float* __restrict__ out) {
    __shared__ float tile[32][33];
    int nb = blockIdx.x, mb = blockIdx.y;
    int tx = threadIdx.x & 31, ty = threadIdx.x >> 5;
#pragma unroll
    for (int i = 0; i < 4; ++i)
        tile[ty + i * 8][tx] = x[(size_t)(nb * 32 + ty + i * 8) * EMB + mb * 32 + tx];
    __syncthreads();
#pragma unroll
    for (int i = 0; i < 4; ++i)
        out[(size_t)(mb * 32 + ty + i * 8) * NTOK + nb * 32 + tx] =
            fmaxf(tile[tx][ty + i * 8], 0.0f);
}

// ---------------------------------------------------------------------------
// Host orchestration
// ---------------------------------------------------------------------------
extern "C" void kernel_launch(void* const* d_in, const int* in_sizes, int n_in,
                              void* d_out, int out_size, void* d_ws, size_t ws_size,
                              hipStream_t stream) {
    const float* img   = (const float*)d_in[0];
    const float* ln1_g = (const float*)d_in[1];
    const float* ln1_b = (const float*)d_in[2];
    const float* Wqkv  = (const float*)d_in[3];
    const float* bqkv  = (const float*)d_in[4];
    const float* Wproj = (const float*)d_in[5];
    const float* bproj = (const float*)d_in[6];
    const float* ln2_g = (const float*)d_in[7];
    const float* ln2_b = (const float*)d_in[8];
    const float* W1    = (const float*)d_in[9];
    const float* b1    = (const float*)d_in[10];
    const float* W2    = (const float*)d_in[11];
    const float* b2    = (const float*)d_in[12];
    float* out = (float*)d_out;

    char* ws = (char*)d_ws;
    // workspace layout (bytes):
    //   xbuf  f32 [N,128]                       : 67,108,864
    //   xn    f16 [N,128]                       : 33,554,432
    //   R     (shared region, 134,217,728 B):
    //     phase A: qkv f16 [N,384] (first 100,663,296) + attnout f16 [N,128] (tail)
    //     phase B: h   f16 [N,512] (whole region)
    //   wfmt  f16 (3 layers x 196,608 halves)   : 1,179,648
    //   bqp   f32 (3 x 384)                     : 4,608
    //   stats f32 (256)                         : 1,024
    float* xbuf   = (float*)(ws);
    f16*   xn     = (f16*)(ws + 67108864);
    char*  R      = ws + 100663296;
    f16*   qkvb   = (f16*)(R);
    f16*   attnout= (f16*)(R + 100663296);
    f16*   hbuf   = (f16*)(R);
    f16*   wfmt   = (f16*)(ws + 234881024);
    float* bqp    = (float*)(ws + 236060672);
    float* stats  = (float*)(ws + 236065280);

    const size_t WQKV_OFF = 0, WPROJ_OFF = 49152, W1_OFF = 65536, W2_OFF = 131072;
    const size_t WPL = 196608;  // halves per layer

    // ---- pre-format all weights (tiny, once per call) ----
    for (int L = 0; L < 3; ++L) {
        preformat_w<<<192, 256, 0, stream>>>(Wqkv + (size_t)L * 128 * 384,
                                             wfmt + L * WPL + WQKV_OFF, 128, 384, 1);
        preformat_w<<<64, 256, 0, stream>>>(Wproj + (size_t)L * 128 * 128,
                                            wfmt + L * WPL + WPROJ_OFF, 128, 128, 0);
        preformat_w<<<256, 256, 0, stream>>>(W1 + (size_t)L * 128 * 512,
                                             wfmt + L * WPL + W1_OFF, 128, 512, 0);
        preformat_w<<<256, 256, 0, stream>>>(W2 + (size_t)L * 512 * 128,
                                             wfmt + L * WPL + W2_OFF, 512, 128, 0);
        permute_bqkv<<<3, 128, 0, stream>>>(bqkv + (size_t)L * 384, bqp + L * 384);
    }

    // ---- img [1,128,Z,Y,X] -> xbuf [N,128] ----
    transpose_in<<<dim3(NTOK / 32, 4), 256, 0, stream>>>(img, xbuf);

    const int MT = NTOK / 16;  // 8192 row tiles
    for (int L = 0; L < 3; ++L) {
        const f16* wq = wfmt + L * WPL + WQKV_OFF;
        const f16* wp = wfmt + L * WPL + WPROJ_OFF;
        const f16* w1 = wfmt + L * WPL + W1_OFF;
        const f16* w2 = wfmt + L * WPL + W2_OFF;

        // LN1 -> xn (f16)
        zero_stats<<<1, 256, 0, stream>>>(stats);
        stats_kernel<<<1024, 128, 0, stream>>>(xbuf, stats);
        normcast_kernel<<<NTOK * EMB / 256, 256, 0, stream>>>(
            xbuf, stats, ln1_g + L * 128, ln1_b + L * 128, xn);

        // QKV: [N,128] x [128,384] -> qkv f16 (de-interleaved q|k|v)
        gemm_kernel<0, 0><<<MT * 6 / 8, 256, 0, stream>>>(xn, wq, bqp + L * 384,
                                                          qkvb, nullptr, 128, 384);
        // attention per head -> attnout f16 [N,128]
        attn_kernel<<<XY / 8, 256, 0, stream>>>(qkvb, attnout);

        // proj + residual into xbuf
        gemm_kernel<0, 1><<<MT * 2 / 8, 256, 0, stream>>>(attnout, wp, bproj + L * 128,
                                                          nullptr, xbuf, 128, 128);

        // LN2 -> xn (f16)
        zero_stats<<<1, 256, 0, stream>>>(stats);
        stats_kernel<<<1024, 128, 0, stream>>>(xbuf, stats);
        normcast_kernel<<<NTOK * EMB / 256, 256, 0, stream>>>(
            xbuf, stats, ln2_g + L * 128, ln2_b + L * 128, xn);

        // FFN1 (+hard_swish) -> h f16 [N,512]
        gemm_kernel<1, 0><<<MT * 8 / 8, 256, 0, stream>>>(xn, w1, b1 + L * 512,
                                                          hbuf, nullptr, 128, 512);
        // FFN2 + residual into xbuf
        gemm_kernel<0, 1><<<MT * 2 / 8, 256, 0, stream>>>(hbuf, w2, b2 + L * 128,
                                                          nullptr, xbuf, 512, 128);
    }

    // ---- relu + transpose back to [1,128,Z,Y,X] ----
    final_out<<<dim3(NTOK / 32, 4), 256, 0, stream>>>(xbuf, out);
    (void)in_sizes; (void)n_in; (void)out_size; (void)ws_size;
}
